// GIN_1005022347909
// MI455X (gfx1250) — compile-verified, matrix operand order checked by reading, not run on hardware
//
#include <hip/hip_runtime.h>

typedef __attribute__((ext_vector_type(16))) __bf16 v16bf;
typedef __attribute__((ext_vector_type(8)))  float  v8f;

#define N_NODES 100000
#define N_EDGES 1600000
#define DHID    128

// ---------------------------------------------------------------------------
// Weight prep: WT[l][n][k] = W[l][k][n] in bf16 (column-major B == row-major BT)
// lin_w padded from 40 to 48 output columns with zeros.
// ---------------------------------------------------------------------------
__global__ void prep_weights(const float* __restrict__ W1, const float* __restrict__ W2,
                             const float* __restrict__ linw,
                             __bf16* __restrict__ WT1, __bf16* __restrict__ WT2,
                             __bf16* __restrict__ linT)
{
    int idx = blockIdx.x * blockDim.x + threadIdx.x;
    const int perW = 3 * DHID * DHID;                 // 49152
    if (idx < perW) {
        int l = idx >> 14;
        int n = (idx >> 7) & 127;
        int k = idx & 127;
        WT1[idx] = (__bf16)W1[(l << 14) + k * DHID + n];
        WT2[idx] = (__bf16)W2[(l << 14) + k * DHID + n];
    }
    if (idx < 48 * DHID) {
        int n = idx >> 7;
        int k = idx & 127;
        linT[idx] = (n < 40) ? (__bf16)linw[k * 40 + n] : (__bf16)0.0f;
    }
}

// ---------------------------------------------------------------------------
// agg = x  (the (1+eps)*x term, eps=0)
// ---------------------------------------------------------------------------
__global__ void copy_f32x4(const float4* __restrict__ src, float4* __restrict__ dst, int n4)
{
    int i = blockIdx.x * blockDim.x + threadIdx.x;
    if (i < n4) dst[i] = src[i];
}

// ---------------------------------------------------------------------------
// One wave per edge: gather x[src] (lane holds 4 contiguous f32),
// scatter-add into agg[dst] with hardware f32 atomics (L2-resident).
// ---------------------------------------------------------------------------
__global__ void scatter_edges(const float* __restrict__ x, float* __restrict__ agg,
                              const long long* __restrict__ src,
                              const long long* __restrict__ dst, int nEdges)
{
    int wid  = (blockIdx.x * blockDim.x + threadIdx.x) >> 5;
    int lane = threadIdx.x & 31;
    if (wid >= nEdges) return;
    long long s = src[wid];
    long long d = dst[wid];
    float4 v = ((const float4*)(x + (size_t)s * DHID))[lane];
    float* ad = agg + (size_t)d * DHID + lane * 4;
    unsafeAtomicAdd(ad + 0, v.x);
    unsafeAtomicAdd(ad + 1, v.y);
    unsafeAtomicAdd(ad + 2, v.z);
    unsafeAtomicAdd(ad + 3, v.w);
}

// ---------------------------------------------------------------------------
// C[16x16 per wave] = relu?(A(f32->bf16) @ BT^T + bias), K = 128.
// Block = 16 rows of A; wave w = N-tile w. 4x v_wmma_f32_16x16x32_bf16.
// A-fragment per-lane layout (ISA 7.12.2, 16-bit A 16x32):
//   lane<16: kb=0 ; lane>=16: kb=8 ; VGPR v<4 -> K=kb+2v ; v>=4 -> K=kb+16+2(v-4)
// C layout: VGPR r, lanes 0-15 -> M=r, lanes 16-31 -> M=8+r; N = lane&15.
// ---------------------------------------------------------------------------
__global__ void gin_gemm(const float* __restrict__ A, const __bf16* __restrict__ BT,
                         const float* __restrict__ bias, float* __restrict__ C,
                         int nOut, int ldc, int doRelu)
{
    const int lane    = threadIdx.x & 31;
    const int wave    = threadIdx.x >> 5;
    const int rowBase = blockIdx.x * 16;
    const int r15     = lane & 15;
    const int hi      = lane >> 4;
    const int kb      = hi * 8;

    const float*  Ap = A  + (size_t)(rowBase + r15)   * DHID;
    const __bf16* Bp = BT + (size_t)(wave * 16 + r15) * DHID;

    v8f acc = {};
    #pragma unroll
    for (int kc = 0; kc < DHID; kc += 32) {
        v16bf a, b;
        #pragma unroll
        for (int v = 0; v < 8; ++v) {
            const int k0 = kc + kb + ((v < 4) ? (2 * v) : (16 + 2 * (v - 4)));
            float2 f = *(const float2*)(Ap + k0);
            a[2 * v]     = (__bf16)f.x;          // v_cvt to bf16 in-register
            a[2 * v + 1] = (__bf16)f.y;
            b[2 * v]     = Bp[k0];
            b[2 * v + 1] = Bp[k0 + 1];
        }
        acc = __builtin_amdgcn_wmma_f32_16x16x32_bf16(
                  false, a, false, b, (short)0, acc, false, false);
    }

    const int n  = wave * 16 + r15;
    const int mb = hi * 8;
    float bv = (n < nOut) ? bias[n] : 0.0f;
    #pragma unroll
    for (int r = 0; r < 8; ++r) {
        float val = acc[r] + bv;
        if (doRelu) val = fmaxf(val, 0.0f);
        if (n < nOut) C[(size_t)(rowBase + mb + r) * ldc + n] = val;
    }
}

// ---------------------------------------------------------------------------
extern "C" void kernel_launch(void* const* d_in, const int* in_sizes, int n_in,
                              void* d_out, int out_size, void* d_ws, size_t ws_size,
                              hipStream_t stream)
{
    const float*     x    = (const float*)d_in[0];
    const long long* ei   = (const long long*)d_in[1];
    const float*     W1   = (const float*)d_in[2];
    const float*     b1   = (const float*)d_in[3];
    const float*     W2   = (const float*)d_in[4];
    const float*     b2   = (const float*)d_in[5];
    const float*     linw = (const float*)d_in[6];
    const float*     linb = (const float*)d_in[7];
    float* out = (float*)d_out;

    char* ws = (char*)d_ws;
    const size_t nd = (size_t)N_NODES * DHID * sizeof(float);   // 51.2 MB
    float*  agg  = (float*)(ws);
    float*  hbuf = (float*)(ws + nd);
    float*  xbuf = (float*)(ws + 2 * nd);
    __bf16* WT1  = (__bf16*)(ws + 3 * nd);
    __bf16* WT2  = WT1 + 3 * DHID * DHID;
    __bf16* linT = WT2 + 3 * DHID * DHID;

    prep_weights<<<(3 * DHID * DHID + 255) / 256, 256, 0, stream>>>(W1, W2, linw, WT1, WT2, linT);

    const int n4 = N_NODES * DHID / 4;                          // 3.2M float4
    const float* xcur = x;
    for (int l = 0; l < 3; ++l) {
        copy_f32x4<<<(n4 + 255) / 256, 256, 0, stream>>>((const float4*)xcur, (float4*)agg, n4);
        scatter_edges<<<(N_EDGES * 32 + 255) / 256, 256, 0, stream>>>(xcur, agg, ei, ei + N_EDGES, N_EDGES);
        gin_gemm<<<N_NODES / 16, 256, 0, stream>>>(agg,  WT1 + l * DHID * DHID, b1 + l * DHID, hbuf, DHID, DHID, 1);
        gin_gemm<<<N_NODES / 16, 256, 0, stream>>>(hbuf, WT2 + l * DHID * DHID, b2 + l * DHID, xbuf, DHID, DHID, 1);
        xcur = xbuf;
    }
    gin_gemm<<<N_NODES / 16, 96, 0, stream>>>(xbuf, linT, linb, out, 40, 40, 0);
}
